// StyleLoss_23768349016191
// MI455X (gfx1250) — compile-verified, matrix operand order checked by reading, not run on hardware
//
#include <hip/hip_runtime.h>

// ---------------- problem constants ----------------
constexpr int C  = 256, H = 96, W = 96;
constexpr int OH = 94, OW = 94;
constexpr int Q  = OH * OW;          // 8836 synth patches
constexpr int S  = Q;                // 8836 style patches
constexpr int D  = C * 9;            // 2304 patch dim (C, kh, kw ordering)

constexpr int MB = 128;              // Q-rows per block (4 m-waves x 32)
constexpr int NB = 128;              // S-cols per macro tile (2 n-waves x 64)
constexpr int KC = 192;              // K-chunk staged in LDS: 128*192*2 = 48 KB
constexpr int NSPLIT = 8;            // S-range splits (occupancy: 70 x 8 = 560 WGs)
constexpr int NSLOT  = NSPLIT * 2;   // partial argmax slots (x 2 n-waves)

constexpr int QP = 8960;             // Q padded to multiple of MB
constexpr int SP = 8960;             // S padded to multiple of NB

typedef __attribute__((ext_vector_type(16))) __bf16 v16bf;
typedef __attribute__((ext_vector_type(8)))  float  v8f;

static __device__ inline unsigned short f2bf(float f) {
    unsigned int u = __float_as_uint(f);
    u += 0x7FFF + ((u >> 16) & 1);   // round-to-nearest-even
    return (unsigned short)(u >> 16);
}

static __device__ inline v16bf load_frag(const unsigned short* p0,
                                         const unsigned short* p1) {
    union { uint4 u[2]; v16bf v; } t;
    t.u[0] = *(const uint4*)p0;      // K = khalf*8 + 0..7   (VGPR 0-3)
    t.u[1] = *(const uint4*)p1;      // K = 16 + khalf*8 ... (VGPR 4-7)
    return t.v;
}

// ---------------- kernel 1: patch extraction + bf16 convert + norms ----------
__global__ void extract_patches(const float* __restrict__ img,
                                unsigned short* __restrict__ mat,
                                float* __restrict__ norms,
                                int valid, int padded) {
    int wave = (int)((blockIdx.x * blockDim.x + threadIdx.x) >> 5);
    int lane = threadIdx.x & 31;
    if (wave >= padded) return;
    float sumsq = 0.f;
    if (wave < valid) {
        int i = wave / OW, j = wave % OW;
        for (int t = 0; t < D / 32; ++t) {
            int d   = t * 32 + lane;
            int c   = d / 9;
            int rem = d - c * 9;
            int kh  = rem / 3, kw = rem - kh * 3;
            float v = img[c * (H * W) + (i + kh) * W + (j + kw)];
            sumsq += v * v;
            mat[(size_t)wave * D + d] = f2bf(v);
        }
    } else {
        for (int t = 0; t < D / 32; ++t)
            mat[(size_t)wave * D + t * 32 + lane] = 0;
    }
    if (norms) {
        for (int off = 16; off; off >>= 1) sumsq += __shfl_xor(sumsq, off, 32);
        if (lane == 0) norms[wave] = (wave < valid) ? sqrtf(sumsq) : 1.0f;
    }
}

// ---------------- kernel 2: bf16 WMMA GEMM fused with partial argmax ---------
// Block: 256 threads = 8 waves = 4 m-waves (32 q-rows each, 2 M-tiles) x
// 2 n-waves (64 s-cols each, 4 N-tiles). Macro tile: 128 q-rows x 128 s-cols.
// LDS stages a 128-row x 192-col style K-chunk, filled with the gfx1250
// async direct-to-LDS path (GLOBAL_LOAD_ASYNC_TO_LDS_B128 + ASYNCcnt).
// Per k-step: 2 A frags (4 global b128) + 4 B frags (8 ds b128) feed
// 8 independent WMMAs (1 ds_load / 0.5 global per wmma).
__global__ __launch_bounds__(256)
void gemm_argmax_part(const unsigned short* __restrict__ A,   // synth [QP][D]
                      const unsigned short* __restrict__ B,   // style [SP][D]
                      const float* __restrict__ norms,        // [SP]
                      float* __restrict__ pval,               // [NSLOT][QP]
                      int*   __restrict__ pidx) {             // [NSLOT][QP]
    __shared__ unsigned short Bs[NB * KC];                    // 48 KB
    const int tid  = threadIdx.x;
    const int wave = tid >> 5;
    const int lane = tid & 31;
    const int h    = lane >> 4;      // K-half selector (A/B fragment layout)
    const int n    = lane & 15;
    const int mw   = wave & 3;       // m-wave: which 32 q-rows
    const int nw   = wave >> 2;      // n-wave: which 64 s-cols
    const int qtile = blockIdx.x * MB + mw * 32;
    const int sy    = blockIdx.y;

    const unsigned short* arow0 = A + (size_t)(qtile + n) * D;
    const unsigned short* arow1 = A + (size_t)(qtile + 16 + n) * D;

    // LDS byte address of Bs (generic->LDS: low 32 bits are the LDS offset)
    const unsigned ldsbase = (unsigned)(uintptr_t)(void*)Bs;

    float bestv[2][8];
    int   besti[2][8];
#pragma unroll
    for (int mt = 0; mt < 2; ++mt)
#pragma unroll
        for (int r = 0; r < 8; ++r) { bestv[mt][r] = -3.4e38f; besti[mt][r] = 0; }

    for (int t = sy; t < SP / NB; t += NSPLIT) {
        const int sbase = t * NB;

        v8f acc[2][4] = {{{}, {}, {}, {}}, {{}, {}, {}, {}}};
        for (int kc = 0; kc < D; kc += KC) {
            __syncthreads();  // protect previous chunk reads
            // async stage: style rows [sbase, sbase+128) x K [kc, kc+KC)
            // straight into LDS, no VGPR round-trip (tracked by ASYNCcnt)
#pragma unroll
            for (int u = 0; u < 12; ++u) {
                int idx = tid + u * 256;           // 0..3071 (16B units)
                int r   = idx / (KC / 8);
                int ci  = idx - r * (KC / 8);
                const void* src =
                    (const void*)(B + (size_t)(sbase + r) * D + kc + ci * 8);
                unsigned dst = ldsbase + (unsigned)idx * 16u;
                asm volatile("global_load_async_to_lds_b128 %0, %1, off"
                             :: "v"(dst), "v"(src) : "memory");
            }
            asm volatile("s_wait_asynccnt 0" ::: "memory");
            __syncthreads();

#pragma unroll
            for (int kl = 0; kl < KC; kl += 32) {
                // distinct live fragments so the scheduler can overlap waits
                v16bf a0 = load_frag(arow0 + kc + kl + h * 8,
                                     arow0 + kc + kl + 16 + h * 8);
                v16bf a1 = load_frag(arow1 + kc + kl + h * 8,
                                     arow1 + kc + kl + 16 + h * 8);
                const unsigned short* bbase = Bs + (size_t)(nw * 64 + n) * KC + kl;
                v16bf b0 = load_frag(bbase + h * 8,            bbase + 16 + h * 8);
                v16bf b1 = load_frag(bbase + 16 * KC + h * 8,  bbase + 16 * KC + 16 + h * 8);
                v16bf b2 = load_frag(bbase + 32 * KC + h * 8,  bbase + 32 * KC + 16 + h * 8);
                v16bf b3 = load_frag(bbase + 48 * KC + h * 8,  bbase + 48 * KC + 16 + h * 8);
                acc[0][0] = __builtin_amdgcn_wmma_f32_16x16x32_bf16(
                    false, a0, false, b0, (short)0, acc[0][0], false, false);
                acc[0][1] = __builtin_amdgcn_wmma_f32_16x16x32_bf16(
                    false, a0, false, b1, (short)0, acc[0][1], false, false);
                acc[0][2] = __builtin_amdgcn_wmma_f32_16x16x32_bf16(
                    false, a0, false, b2, (short)0, acc[0][2], false, false);
                acc[0][3] = __builtin_amdgcn_wmma_f32_16x16x32_bf16(
                    false, a0, false, b3, (short)0, acc[0][3], false, false);
                acc[1][0] = __builtin_amdgcn_wmma_f32_16x16x32_bf16(
                    false, a1, false, b0, (short)0, acc[1][0], false, false);
                acc[1][1] = __builtin_amdgcn_wmma_f32_16x16x32_bf16(
                    false, a1, false, b1, (short)0, acc[1][1], false, false);
                acc[1][2] = __builtin_amdgcn_wmma_f32_16x16x32_bf16(
                    false, a1, false, b2, (short)0, acc[1][2], false, false);
                acc[1][3] = __builtin_amdgcn_wmma_f32_16x16x32_bf16(
                    false, a1, false, b3, (short)0, acc[1][3], false, false);
            }
        }

        // one reciprocal per column group (off the WMMA critical path)
        float rinv[4];
        int   cols[4];
#pragma unroll
        for (int g = 0; g < 4; ++g) {
            cols[g] = sbase + nw * 64 + g * 16 + n;
            rinv[g] = 1.0f / norms[cols[g]];
        }

        // argmax: per-lane over the 4 groups, then 16-lane butterfly.
        // C layout: lane holds (m = r + 8*h, col group g) in acc[mt][g][r].
#pragma unroll
        for (int mt = 0; mt < 2; ++mt) {
#pragma unroll
            for (int r = 0; r < 8; ++r) {
                float v = -3.4e38f;
                int idx = 0;
#pragma unroll
                for (int g = 0; g < 4; ++g) {
                    float nv = (cols[g] < S) ? acc[mt][g][r] * rinv[g] : -3.4e38f;
                    if (nv > v) { v = nv; idx = cols[g]; }  // ascending g => lowest idx on ties
                }
#pragma unroll
                for (int off = 1; off < 16; off <<= 1) {
                    float ov = __shfl_xor(v, off, 32);
                    int   oi = __shfl_xor(idx, off, 32);
                    if (ov > v || (ov == v && oi < idx)) { v = ov; idx = oi; }
                }
                if (v > bestv[mt][r] || (v == bestv[mt][r] && idx < besti[mt][r])) {
                    bestv[mt][r] = v; besti[mt][r] = idx;   // tiles ascend within split
                }
            }
        }
    }

    if (n == 0) {
        const int slot = sy * 2 + nw;
#pragma unroll
        for (int mt = 0; mt < 2; ++mt) {
#pragma unroll
            for (int r = 0; r < 8; ++r) {
                int q = qtile + mt * 16 + h * 8 + r;
                if (q < Q) {
                    pval[(size_t)slot * QP + q] = bestv[mt][r];
                    pidx[(size_t)slot * QP + q] = besti[mt][r];
                }
            }
        }
    }
}

// ---------------- kernel 3: deterministic argmax merge across slots ----------
__global__ void merge_nn(const float* __restrict__ pval,
                         const int* __restrict__ pidx,
                         int* __restrict__ nn) {
    int q = blockIdx.x * 256 + threadIdx.x;
    if (q >= Q) return;
    float bv = -3.4e38f;
    int   bi = 0x7fffffff;
    for (int s2 = 0; s2 < NSLOT; ++s2) {
        float v = pval[(size_t)s2 * QP + q];
        int   i = pidx[(size_t)s2 * QP + q];
        if (v > bv || (v == bv && i < bi)) { bv = v; bi = i; }
    }
    nn[q] = bi;
}

// ---------------- kernel 4: per-q gathered MSE (f32, from original images) ---
__global__ void loss_partial(const float* __restrict__ input,
                             const float* __restrict__ target,
                             const int* __restrict__ nn,
                             float* __restrict__ qpart) {
    int wave = (int)((blockIdx.x * blockDim.x + threadIdx.x) >> 5);
    int lane = threadIdx.x & 31;
    if (wave >= Q) return;
    int i = wave / OW, j = wave % OW;
    int s = nn[wave];
    int si = s / OW, sj = s % OW;
    float sum = 0.f;
    for (int t = 0; t < D / 32; ++t) {
        int d   = t * 32 + lane;
        int c   = d / 9;
        int rem = d - c * 9;
        int kh  = rem / 3, kw = rem - kh * 3;
        float a = input[c * (H * W) + (i + kh) * W + (j + kw)];
        float b = target[c * (H * W) + (si + kh) * W + (sj + kw)];
        float df = a - b;
        sum += df * df;
    }
    for (int off = 16; off; off >>= 1) sum += __shfl_xor(sum, off, 32);
    if (lane == 0) qpart[wave] = sum * (1.0f / D);
}

// ---------------- kernel 5: deterministic final reduction --------------------
__global__ void final_reduce(const float* __restrict__ qpart,
                             float* __restrict__ out) {
    __shared__ float sm[256];
    float s = 0.f;
    for (int t = threadIdx.x; t < Q; t += 256) s += qpart[t];
    sm[threadIdx.x] = s;
    __syncthreads();
    for (int off = 128; off; off >>= 1) {
        if ((int)threadIdx.x < off) sm[threadIdx.x] += sm[threadIdx.x + off];
        __syncthreads();
    }
    if (threadIdx.x == 0) out[0] = sm[0] * (1.0f / Q);
}

// ---------------- launch -----------------------------------------------------
extern "C" void kernel_launch(void* const* d_in, const int* in_sizes, int n_in,
                              void* d_out, int out_size, void* d_ws, size_t ws_size,
                              hipStream_t stream) {
    const float* input  = (const float*)d_in[0];
    const float* target = (const float*)d_in[1];

    char* ws = (char*)d_ws;
    size_t off = 0;
    auto alloc = [&](size_t bytes) -> void* {
        void* p = ws + off;
        off = (off + bytes + 255) & ~(size_t)255;
        return p;
    };
    unsigned short* synthB = (unsigned short*)alloc((size_t)QP * D * 2); // 41.3 MB
    unsigned short* styleB = (unsigned short*)alloc((size_t)SP * D * 2); // 41.3 MB
    float* norms = (float*)alloc((size_t)SP * 4);
    float* pval  = (float*)alloc((size_t)NSLOT * QP * 4);
    int*   pidx  = (int*)alloc((size_t)NSLOT * QP * 4);
    int*   nn    = (int*)alloc((size_t)Q * 4);
    float* qpart = (float*)alloc((size_t)Q * 4);

    extract_patches<<<QP / 8, 256, 0, stream>>>(input,  synthB, nullptr, Q, QP);
    extract_patches<<<SP / 8, 256, 0, stream>>>(target, styleB, norms,   S, SP);

    dim3 grid(QP / MB, NSPLIT);      // 70 x 8 = 560 workgroups
    gemm_argmax_part<<<grid, 256, 0, stream>>>(synthB, styleB, norms, pval, pidx);

    merge_nn<<<(Q + 255) / 256, 256, 0, stream>>>(pval, pidx, nn);
    loss_partial<<<(Q + 7) / 8, 256, 0, stream>>>(input, target, nn, qpart);
    final_reduce<<<1, 256, 0, stream>>>(qpart, (float*)d_out);
}
